// RNN_57492432224676
// MI455X (gfx1250) — matrix operand order repaired and checked by reading, not compiled, not run on hardware
//
#include <hip/hip_runtime.h>
#include <cstdint>

typedef __attribute__((ext_vector_type(16))) _Float16 v16h;
typedef __attribute__((ext_vector_type(8)))  _Float16 v8h;
typedef __attribute__((ext_vector_type(8)))  float    v8f;

#define SEQ     22
#define HIDDEN  128
#define IN_SZ   1000
#define KPAD    1024
#define BATCH   2048
#define GATES   384                 // 3*HIDDEN
#define BT      (BATCH * SEQ)       // 45056 rows for the input GEMM
#define DEC_K   (SEQ * HIDDEN)      // 2816

// ---------------- CDNA5 async copy helpers ----------------

// Direct global -> LDS async copy (ASYNCcnt), bypasses VGPRs.
__device__ __forceinline__ void async_copy_b128(uint32_t lds_addr, const void* gaddr) {
    asm volatile("global_load_async_to_lds_b128 %0, %1, off"
                 :: "v"(lds_addr), "v"(gaddr)
                 : "memory");
}

__device__ __forceinline__ void wait_asynccnt0() {
    asm volatile("s_wait_asynccnt 0" ::: "memory");
}

// ---------------- conversion kernels ----------------

// f32 [rows x srcK] -> f16 [rows x 1024], zero padded in K
__global__ void cvt_pad1024(const float* __restrict__ src,
                            _Float16* __restrict__ dst,
                            int rows, int srcK) {
    int idx = blockIdx.x * blockDim.x + threadIdx.x;
    int total = rows << 10;
    if (idx >= total) return;
    int r = idx >> 10;
    int k = idx & 1023;
    dst[idx] = (k < srcK) ? (_Float16)src[(size_t)r * srcK + k] : (_Float16)0.0f;
}

__global__ void cvt_plain(const float* __restrict__ src,
                          _Float16* __restrict__ dst, int n) {
    int idx = blockIdx.x * blockDim.x + threadIdx.x;
    if (idx < n) dst[idx] = (_Float16)src[idx];
}

// ---------------- WMMA GEMM with async double buffering ----------------
// C[M x N] = A[M x K] * B[N x K]^T + bias  (A,B f16, C f32)
// Block: 256 threads (8 waves). Block tile 128(M) x 64(N).
// Wave w computes rows [m0+16w, m0+16w+16) x 64 cols -> 4 wmma accum tiles.
// Tiles for step s+1 are async-copied into the other LDS buffer while the
// WMMAs of step s run.
__global__ __launch_bounds__(256)
void wmma_gemm(const _Float16* __restrict__ A, int lda,
               const _Float16* __restrict__ B, int ldb,
               float* __restrict__ C, int ldc,
               const float* __restrict__ bias, int K) {
    __shared__ _Float16 As[2][128][40];   // 32 K + 8 pad, double buffered
    __shared__ _Float16 Bs[2][64][40];    // [col][K] layout, double buffered

    const int tid  = threadIdx.x;
    const int lane = tid & 31;
    const int wave = tid >> 5;     // 0..7 -> M strip within block tile
    const int half = lane >> 4;    // 0/1
    const int l16  = lane & 15;
    const int m0 = blockIdx.x * 128;
    const int n0 = blockIdx.y * 64;

    // Per-thread staging: A chunks (tid) and (tid+256), B chunk (tid).
    // chunk ch: row = ch>>2, 8-half column = (ch&3)*8.
    const int sr  = tid >> 2;
    const int sc8 = (tid & 3) << 3;
    const _Float16* gA0 = A + (size_t)(m0 + sr) * lda + sc8;
    const _Float16* gA1 = gA0 + (size_t)64 * lda;          // rows 64..127
    const _Float16* gB0 = B + (size_t)(n0 + sr) * ldb + sc8;  // sr < 64 for B

    uint32_t ldsA0[2], ldsA1[2], ldsB0[2];
#pragma unroll
    for (int buf = 0; buf < 2; ++buf) {
        ldsA0[buf] = (uint32_t)(size_t)&As[buf][sr][sc8];
        ldsA1[buf] = (uint32_t)(size_t)&As[buf][sr + 64][sc8];
        ldsB0[buf] = (uint32_t)(size_t)&Bs[buf][sr][sc8];
    }

    v8f acc[4] = {};

    const int nsteps = K >> 5;

    // Prime buffer 0 with tile k0 = 0.
    async_copy_b128(ldsA0[0], gA0);
    async_copy_b128(ldsA1[0], gA1);
    async_copy_b128(ldsB0[0], gB0);

    for (int s = 0; s < nsteps; ++s) {
        wait_asynccnt0();       // my copies into buf[cur] are done
        __syncthreads();        // everyone's copies done; prev readers of nxt done

        if (s + 1 < nsteps) {
            const int nxt = (s + 1) & 1;
            const int koff = (s + 1) << 5;
            async_copy_b128(ldsA0[nxt], gA0 + koff);
            async_copy_b128(ldsA1[nxt], gA1 + koff);
            async_copy_b128(ldsB0[nxt], gB0 + koff);
        }

        const int cur = s & 1;

        // A fragment per ISA layout: lanes 0-15 row m, elems0-7=K0-7, 8-15=K16-23;
        // lanes 16-31: K8-15 / K24-31.
        const int ar = wave * 16 + l16;
        v8h a0 = *(const v8h*)&As[cur][ar][half * 8];
        v8h a1 = *(const v8h*)&As[cur][ar][16 + half * 8];
        v16h af = __builtin_shufflevector(a0, a1,
                    0,1,2,3,4,5,6,7,8,9,10,11,12,13,14,15);

        // Preload all 4 B fragments, then issue WMMAs back-to-back.
        v16h bf[4];
#pragma unroll
        for (int nb = 0; nb < 4; ++nb) {
            const int bc = nb * 16 + l16;
            v8h b0 = *(const v8h*)&Bs[cur][bc][half * 8];
            v8h b1 = *(const v8h*)&Bs[cur][bc][16 + half * 8];
            bf[nb] = __builtin_shufflevector(b0, b1,
                        0,1,2,3,4,5,6,7,8,9,10,11,12,13,14,15);
        }
#pragma unroll
        for (int nb = 0; nb < 4; ++nb) {
            acc[nb] = __builtin_amdgcn_wmma_f32_16x16x32_f16(
                          false, af, false, bf[nb], (short)0, acc[nb], false, false);
        }
    }

    // Epilogue: C/D layout — lanes0-15: VGPR i -> M=i; lanes16-31: M=8+i; N=l16.
    const int mbase = m0 + wave * 16 + half * 8;
#pragma unroll
    for (int nb = 0; nb < 4; ++nb) {
        const int col = n0 + nb * 16 + l16;
        const float bv = bias[col];
#pragma unroll
        for (int i = 0; i < 8; ++i)
            C[(size_t)(mbase + i) * ldc + col] = acc[nb][i] + bv;
    }
}

// ---------------- GRU pointwise update ----------------
__device__ __forceinline__ float sigmoid_f(float x) {
    return 1.0f / (1.0f + __expf(-x));
}

__global__ void gru_pointwise(const float* __restrict__ xg,   // [BT x 384]
                              const float* __restrict__ hg,   // [BATCH x 384]
                              const float* __restrict__ hprev, int hprev_stride,
                              float* __restrict__ hseq,       // [BATCH x SEQ x H]
                              _Float16* __restrict__ hf16,    // [BATCH x H]
                              int t) {
    int idx = blockIdx.x * blockDim.x + threadIdx.x;
    if (idx >= BATCH * HIDDEN) return;
    int b = idx >> 7;
    int j = idx & 127;
    const float* x = xg + (size_t)(b * SEQ + t) * GATES;
    const float* h = hg + (size_t)b * GATES;
    float hp = hprev[(size_t)b * hprev_stride + j];
    float r = sigmoid_f(x[j]       + h[j]);
    float z = sigmoid_f(x[128 + j] + h[128 + j]);
    float n = tanhf(x[256 + j] + r * h[256 + j]);
    float hn = (1.0f - z) * n + z * hp;
    hseq[(size_t)b * DEC_K + t * HIDDEN + j] = hn;
    hf16[idx] = (_Float16)hn;
}

// ---------------- decoder ----------------
__global__ void decoder_kernel(const float* __restrict__ hseq,
                               const float* __restrict__ Wd,   // [4 x 2816]
                               const float* __restrict__ bd,
                               float* __restrict__ out) {
    int idx = blockIdx.x * blockDim.x + threadIdx.x;
    if (idx >= BATCH * 4) return;
    int b = idx >> 2;
    int c = idx & 3;
    const float4* hrow = (const float4*)(hseq + (size_t)b * DEC_K);
    const float4* wrow = (const float4*)(Wd + (size_t)c * DEC_K);
    float s = bd[c];
    for (int k = 0; k < DEC_K / 4; ++k) {
        float4 hv = hrow[k], wv = wrow[k];
        s += hv.x * wv.x + hv.y * wv.y + hv.z * wv.z + hv.w * wv.w;
    }
    out[idx] = s;
}

// ---------------- host orchestration ----------------
extern "C" void kernel_launch(void* const* d_in, const int* in_sizes, int n_in,
                              void* d_out, int out_size, void* d_ws, size_t ws_size,
                              hipStream_t stream) {
    const float* inputs = (const float*)d_in[0];  // [B, T, 1000]
    const float* h0     = (const float*)d_in[1];  // [1, B, 128]
    const float* W_ih   = (const float*)d_in[2];  // [384, 1000]
    const float* W_hh   = (const float*)d_in[3];  // [384, 128]
    const float* b_ih   = (const float*)d_in[4];  // [384]
    const float* b_hh   = (const float*)d_in[5];  // [384]
    const float* W_dec  = (const float*)d_in[6];  // [4, 2816]
    const float* b_dec  = (const float*)d_in[7];  // [4]
    float* out = (float*)d_out;

    char* ws = (char*)d_ws;
    _Float16* A16   = (_Float16*)ws; ws += (size_t)BT * KPAD * 2;        // 92.3 MB
    _Float16* Wih16 = (_Float16*)ws; ws += (size_t)GATES * KPAD * 2;     // 0.8 MB
    _Float16* Whh16 = (_Float16*)ws; ws += (size_t)GATES * HIDDEN * 2;   // 96 KB
    _Float16* H16   = (_Float16*)ws; ws += (size_t)BATCH * HIDDEN * 2;   // 0.5 MB
    float* xg   = (float*)ws;        ws += (size_t)BT * GATES * 4;       // 69.2 MB
    float* hg   = (float*)ws;        ws += (size_t)BATCH * GATES * 4;    // 3.1 MB
    float* hseq = (float*)ws;        ws += (size_t)BATCH * DEC_K * 4;    // 23.1 MB

    // f32 -> f16 staging (zero-pad K to 1024 for the input GEMM)
    {
        int n = BT * KPAD;
        cvt_pad1024<<<(n + 255) / 256, 256, 0, stream>>>(inputs, A16, BT, IN_SZ);
    }
    {
        int n = GATES * KPAD;
        cvt_pad1024<<<(n + 255) / 256, 256, 0, stream>>>(W_ih, Wih16, GATES, IN_SZ);
    }
    {
        int n = GATES * HIDDEN;
        cvt_plain<<<(n + 255) / 256, 256, 0, stream>>>(W_hh, Whh16, n);
    }
    {
        int n = BATCH * HIDDEN;
        cvt_plain<<<(n + 255) / 256, 256, 0, stream>>>(h0, H16, n);
    }

    // x_gates = inputs @ W_ih^T + b_ih   -> [45056 x 384]
    wmma_gemm<<<dim3(BT / 128, GATES / 64), 256, 0, stream>>>(
        A16, KPAD, Wih16, KPAD, xg, GATES, b_ih, KPAD);

    // 22 sequential GRU steps: hg = h @ W_hh^T + b_hh, then gate update
    for (int t = 0; t < SEQ; ++t) {
        wmma_gemm<<<dim3(BATCH / 128, GATES / 64), 256, 0, stream>>>(
            H16, HIDDEN, Whh16, HIDDEN, hg, GATES, b_hh, HIDDEN);
        const float* hprev = (t == 0) ? h0 : (hseq + (size_t)(t - 1) * HIDDEN);
        int hstride = (t == 0) ? HIDDEN : DEC_K;
        int n = BATCH * HIDDEN;
        gru_pointwise<<<(n + 255) / 256, 256, 0, stream>>>(
            xg, hg, hprev, hstride, hseq, H16, t);
    }

    // decoder: out = hseq @ W_dec^T + b_dec  -> [2048 x 4]
    {
        int n = BATCH * 4;
        decoder_kernel<<<(n + 255) / 256, 256, 0, stream>>>(hseq, W_dec, b_dec, out);
    }
}
